// Actor_76965813944959
// MI455X (gfx1250) — compile-verified, matrix operand order checked by reading, not run on hardware
//
#include <hip/hip_runtime.h>
#include <hip/hip_bf16.h>
#include <math.h>

// ---------------------------------------------------------------------------
// GCN actor forward for MI455X (gfx1250, wave32).
//   xw = x @ conv_w          : fp32 WMMA 16x16x4 (native f32 matrix path)
//   h  = conv_b + dinv^2*xw  : self-loop folded into init
//   h += norm * xw[src] @dst : one wave per edge, float4 atomics (L2-resident)
//   LN + pool                : wave-per-node shuffle reduction
//   head                     : single-block MLP + tanh
// ---------------------------------------------------------------------------

#define D_H   128
#define D_A   64
#define TILE  16
#define LN_EPS 1e-5f

typedef __attribute__((ext_vector_type(2)))  float    v2f;
typedef __attribute__((ext_vector_type(8)))  float    v8f;
typedef __attribute__((ext_vector_type(16))) _Float16 v16h;

// ---------------- degree / norm ----------------
__global__ void deg_init_kernel(float* __restrict__ deg, int n) {
  int i = blockIdx.x * blockDim.x + threadIdx.x;
  if (i < n) deg[i] = 1.0f;  // self-loop contribution
}

__global__ void deg_count_kernel(const int* __restrict__ dst, float* __restrict__ deg, int e) {
  int i = blockIdx.x * blockDim.x + threadIdx.x;
  if (i < e) atomicAdd(&deg[dst[i]], 1.0f);
}

__global__ void dinv_kernel(const float* __restrict__ deg, float* __restrict__ dinv, int n) {
  int i = blockIdx.x * blockDim.x + threadIdx.x;
  if (i < n) dinv[i] = rsqrtf(deg[i]);  // deg >= 1 always (self loops)
}

// ---------------- xw = x @ conv_w via WMMA ----------------
// Block = 256 threads = 8 waves. Each block computes a 16-row strip of xw.
// A tile (16x128 fp32, 8KB) staged in LDS; wave w owns output columns
// [16w, 16w+16). K-loop: 32 iterations of V_WMMA_F32_16X16X4_F32.
__global__ __launch_bounds__(256) void gemm_xw_kernel(const float* __restrict__ x,
                                                      const float* __restrict__ w,
                                                      float* __restrict__ xw, int n) {
  __shared__ float tileA[TILE * D_H];
  const int row0 = blockIdx.x * TILE;
  const int tid  = threadIdx.x;

  if (row0 + TILE <= n) {  // fast cooperative float4 load (2048 floats)
    const float4* gsrc = reinterpret_cast<const float4*>(x + (size_t)row0 * D_H);
    float4* ldst = reinterpret_cast<float4*>(tileA);
    ldst[tid]       = gsrc[tid];
    ldst[tid + 256] = gsrc[tid + 256];
  } else {
    for (int i = tid; i < TILE * D_H; i += 256) {
      int rr = row0 + (i >> 7);
      tileA[i] = (rr < n) ? x[(size_t)rr * D_H + (i & (D_H - 1))] : 0.0f;
    }
  }
  __syncthreads();

  const int wave = tid >> 5;
  const int lane = tid & 31;
  const int col0 = wave * 16;
  const int nn   = lane & 15;
  v8f acc = {};

#if defined(__has_builtin) && __has_builtin(__builtin_amdgcn_wmma_f32_16x16x4_f32)
  // Native fp32 path: A 16x4 (lanes 0-15: K=k0,k0+1; lanes 16-31: K=k0+2,k0+3)
  const int m  = lane & 15;
  const int kk = (lane >> 4) * 2;
  for (int k0 = 0; k0 < D_H; k0 += 4) {
    v2f a, b;
    a.x = tileA[m * D_H + k0 + kk];
    a.y = tileA[m * D_H + k0 + kk + 1];
    b.x = w[(size_t)(k0 + kk)     * D_H + col0 + nn];
    b.y = w[(size_t)(k0 + kk + 1) * D_H + col0 + nn];
    acc = __builtin_amdgcn_wmma_f32_16x16x4_f32(false, a, false, b,
                                                (short)0, acc, false, false);
  }
#else
  // Fallback: f16 inputs, f32 accumulate (codegen-confirmed builtin).
  const int m = lane & 15;
  for (int k0 = 0; k0 < D_H; k0 += 32) {
    v16h a, b;
    #pragma unroll
    for (int i = 0; i < 16; ++i) {
      int p = i >> 1, odd = i & 1;
      int g = p >> 2, pp = p & 3;
      int k = k0 + g * 16 + ((lane >= 16) ? 8 : 0) + pp * 2 + odd;
      a[i] = (_Float16)tileA[m * D_H + k];
      b[i] = (_Float16)w[(size_t)k * D_H + col0 + nn];
    }
    acc = __builtin_amdgcn_wmma_f32_16x16x32_f16(false, a, false, b,
                                                 (short)0, acc, false, false);
  }
#endif

  // D layout: VGPR r -> row (lane<16 ? r : 8+r), col = lane&15
  const int mbase = (lane < 16) ? 0 : 8;
  const int ncol  = col0 + nn;
  #pragma unroll
  for (int r = 0; r < 8; ++r) {
    int row = row0 + mbase + r;
    if (row < n) xw[(size_t)row * D_H + ncol] = acc[r];
  }
}

// ---------------- h init: conv_b + self-loop term ----------------
__global__ void h_init_kernel(const float* __restrict__ xw, const float* __restrict__ dinv,
                              const float* __restrict__ conv_b, float* __restrict__ h,
                              int n) {
  int idx = blockIdx.x * blockDim.x + threadIdx.x;
  if (idx < n * D_H) {
    int node = idx >> 7;
    int f    = idx & (D_H - 1);
    float di = dinv[node];
    h[idx] = conv_b[f] + di * di * xw[idx];
  }
}

// ---------------- edge scatter: one wave per edge ----------------
// Lane l moves features [4l, 4l+4): a 512B row per edge, float4 gather from
// L2-resident xw, 4 fp32 atomic adds into L2-resident h.
__global__ __launch_bounds__(256) void scatter_kernel(const int* __restrict__ ei,
                                                      const float* __restrict__ xw,
                                                      const float* __restrict__ dinv,
                                                      float* __restrict__ h, int e) {
  int gid  = blockIdx.x * blockDim.x + threadIdx.x;
  int wid  = gid >> 5;
  int lane = gid & 31;
  int nw   = (gridDim.x * blockDim.x) >> 5;
  for (int ed = wid; ed < e; ed += nw) {
    int s = ei[ed];
    int d = ei[e + ed];
    float nrm = dinv[s] * dinv[d];
    float4 v = reinterpret_cast<const float4*>(xw + (size_t)s * D_H)[lane];
    float* hd = h + (size_t)d * D_H + lane * 4;
    atomicAdd(hd + 0, nrm * v.x);
    atomicAdd(hd + 1, nrm * v.y);
    atomicAdd(hd + 2, nrm * v.z);
    atomicAdd(hd + 3, nrm * v.w);
  }
}

// ---------------- zero pooled ----------------
__global__ void zero_kernel(float* __restrict__ p, int n) {
  int i = blockIdx.x * blockDim.x + threadIdx.x;
  if (i < n) p[i] = 0.0f;
}

// ---------------- fused ReLU + LayerNorm + global pool ----------------
// One wave per node (4 feats/lane), shuffle reductions for mean/var,
// LDS partial pool, one global atomic per feature per block.
__global__ __launch_bounds__(256) void ln_pool_kernel(const float* __restrict__ h,
                                                      const float* __restrict__ g,
                                                      const float* __restrict__ b,
                                                      float* __restrict__ pooled, int n) {
  __shared__ float psum[D_H];
  int tid = threadIdx.x;
  if (tid < D_H) psum[tid] = 0.0f;
  __syncthreads();

  int wave = tid >> 5, lane = tid & 31;
  int node = blockIdx.x * 8 + wave;
  if (node < n) {
    float4 v = reinterpret_cast<const float4*>(h + (size_t)node * D_H)[lane];
    v.x = fmaxf(v.x, 0.0f); v.y = fmaxf(v.y, 0.0f);
    v.z = fmaxf(v.z, 0.0f); v.w = fmaxf(v.w, 0.0f);
    float s = v.x + v.y + v.z + v.w;
    #pragma unroll
    for (int off = 16; off > 0; off >>= 1) s += __shfl_xor(s, off, 32);
    float mu = s * (1.0f / D_H);
    float dx0 = v.x - mu, dx1 = v.y - mu, dx2 = v.z - mu, dx3 = v.w - mu;
    float sq = dx0 * dx0 + dx1 * dx1 + dx2 * dx2 + dx3 * dx3;
    #pragma unroll
    for (int off = 16; off > 0; off >>= 1) sq += __shfl_xor(sq, off, 32);
    float rstd = rsqrtf(sq * (1.0f / D_H) + LN_EPS);
    int f = lane * 4;
    atomicAdd(&psum[f + 0], dx0 * rstd * g[f + 0] + b[f + 0]);
    atomicAdd(&psum[f + 1], dx1 * rstd * g[f + 1] + b[f + 1]);
    atomicAdd(&psum[f + 2], dx2 * rstd * g[f + 2] + b[f + 2]);
    atomicAdd(&psum[f + 3], dx3 * rstd * g[f + 3] + b[f + 3]);
  }
  __syncthreads();
  if (tid < D_H) atomicAdd(&pooled[tid], psum[tid]);
}

// ---------------- tiny MLP head ----------------
__global__ __launch_bounds__(128) void head_kernel(const float* __restrict__ pooled,
                                                   const float* __restrict__ w2,
                                                   const float* __restrict__ b2,
                                                   const float* __restrict__ w3,
                                                   const float* __restrict__ b3,
                                                   float* __restrict__ out) {
  __shared__ float a1[D_H];
  int j = threadIdx.x;
  float acc = b2[j];
  for (int i = 0; i < D_H; ++i) acc += pooled[i] * w2[i * D_H + j];
  a1[j] = fmaxf(acc, 0.0f);
  __syncthreads();
  if (j < D_A) {
    float acc2 = b3[j];
    for (int i = 0; i < D_H; ++i) acc2 += a1[i] * w3[i * D_A + j];
    out[j] = tanhf(acc2);  // MAX_ACTION == 1.0
  }
}

// ---------------------------------------------------------------------------
extern "C" void kernel_launch(void* const* d_in, const int* in_sizes, int n_in,
                              void* d_out, int out_size, void* d_ws, size_t ws_size,
                              hipStream_t stream) {
  const float* x      = (const float*)d_in[0];
  const int*   ei     = (const int*)  d_in[1];   // [2,E] flat: row0 = src, row1 = dst
  const float* conv_w = (const float*)d_in[2];
  const float* conv_b = (const float*)d_in[3];
  const float* ln_g   = (const float*)d_in[4];
  const float* ln_b   = (const float*)d_in[5];
  const float* w2     = (const float*)d_in[6];
  const float* b2     = (const float*)d_in[7];
  const float* w3     = (const float*)d_in[8];
  const float* b3     = (const float*)d_in[9];
  float* out = (float*)d_out;

  const int n = in_sizes[0] / D_H;
  const int e = in_sizes[1] / 2;

  // workspace carve-up
  char* p = (char*)d_ws;
  float* xw     = (float*)p; p += sizeof(float) * (size_t)n * D_H;
  float* h      = (float*)p; p += sizeof(float) * (size_t)n * D_H;
  float* deg    = (float*)p; p += sizeof(float) * (size_t)n;
  float* dinv   = (float*)p; p += sizeof(float) * (size_t)n;
  float* pooled = (float*)p;

  const int nt = 256;
  deg_init_kernel <<<(n + nt - 1) / nt, nt, 0, stream>>>(deg, n);
  deg_count_kernel<<<(e + nt - 1) / nt, nt, 0, stream>>>(ei + e, deg, e);
  dinv_kernel     <<<(n + nt - 1) / nt, nt, 0, stream>>>(deg, dinv, n);

  gemm_xw_kernel  <<<(n + TILE - 1) / TILE, 256, 0, stream>>>(x, conv_w, xw, n);

  h_init_kernel   <<<((size_t)n * D_H + nt - 1) / nt, nt, 0, stream>>>(xw, dinv, conv_b, h, n);
  scatter_kernel  <<<2048, 256, 0, stream>>>(ei, xw, dinv, h, e);

  zero_kernel     <<<1, D_H, 0, stream>>>(pooled, D_H);
  ln_pool_kernel  <<<(n + 7) / 8, 256, 0, stream>>>(h, ln_g, ln_b, pooled, n);

  head_kernel     <<<1, D_H, 0, stream>>>(pooled, w2, b2, w3, b3, out);
}
